// LinformerTextEncoder_51642686767111
// MI455X (gfx1250) — matrix operand order repaired and checked by reading, not compiled, not run on hardware
//
#include <hip/hip_runtime.h>
#include <cstdint>

// ---------------- problem constants ----------------
constexpr int Bb = 32, Nn = 2048, Dd = 1024, Hh = 16, Dh = 64, Kk = 64, Ll = 4, Ff = 4096;
constexpr int BN = Bb * Nn;  // 65536 token rows

// ---------------- types ----------------
typedef __attribute__((ext_vector_type(16))) __bf16 v16bf;
typedef __attribute__((ext_vector_type(8)))  float  v8f;
typedef int v4i __attribute__((ext_vector_type(4)));
typedef unsigned int v4u __attribute__((ext_vector_type(4)));
typedef unsigned int v8u __attribute__((ext_vector_type(8)));

#define AS1 __attribute__((address_space(1)))
#define AS3 __attribute__((address_space(3)))

#if __has_builtin(__builtin_amdgcn_global_load_async_to_lds_b128)
#define HAVE_ASYNC_LDS 1
#else
#define HAVE_ASYNC_LDS 0
#endif

__device__ __forceinline__ void wait_asynccnt0() {
#if __has_builtin(__builtin_amdgcn_s_wait_asynccnt)
  __builtin_amdgcn_s_wait_asynccnt(0);
#else
  asm volatile("s_wait_asynccnt 0x0" ::: "memory");
#endif
}

union FragBf {
  v16bf v;
  uint4 q[2];
  unsigned short us[16];
};

__device__ __forceinline__ unsigned short f2bf(float f) {
  unsigned int u = __float_as_uint(f);
  u += 0x7FFFu + ((u >> 16) & 1u);  // round-to-nearest-even
  return (unsigned short)(u >> 16);
}
__device__ __forceinline__ float bf2f(unsigned short b) {
  return __uint_as_float(((unsigned int)b) << 16);
}

// ---------------- fp32 -> bf16 convert (weights) ----------------
__global__ __launch_bounds__(256)
void cvt_f32_bf16_kernel(const float* __restrict__ in, unsigned short* __restrict__ out, int n) {
  for (int i = blockIdx.x * 256 + threadIdx.x; i < n; i += gridDim.x * 256)
    out[i] = f2bf(in[i]);
}

// ---------------- embedding: x = tok[id] + pos ----------------
__global__ __launch_bounds__(256)
void embed_kernel(const int* __restrict__ ids, const float* __restrict__ tok,
                  const float* __restrict__ pos, float* __restrict__ x) {
  const int bn = blockIdx.x;          // [0, BN)
  const int n  = bn & (Nn - 1);
  const int t  = threadIdx.x;
  const long long id = ids[bn];
#pragma unroll
  for (int j = 0; j < 4; ++j) {
    const int i = t + j * 256;
    x[(size_t)bn * Dd + i] = tok[(size_t)id * Dd + i] + pos[(size_t)n * Dd + i];
  }
}

// ---------------- layernorm fp32 -> bf16 ----------------
__global__ __launch_bounds__(256)
void ln_bf16_kernel(const float* __restrict__ X, const float* __restrict__ G,
                    const float* __restrict__ Bv, unsigned short* __restrict__ Out) {
  __shared__ float red[256];
  const int row = blockIdx.x, t = threadIdx.x;
  const float* xr = X + (size_t)row * Dd;
  float v[4];
  float s = 0.f;
#pragma unroll
  for (int j = 0; j < 4; ++j) { v[j] = xr[t + j * 256]; s += v[j]; }
  red[t] = s; __syncthreads();
  for (int st = 128; st > 0; st >>= 1) { if (t < st) red[t] += red[t + st]; __syncthreads(); }
  const float mean = red[0] * (1.0f / (float)Dd);
  __syncthreads();
  float qs = 0.f;
#pragma unroll
  for (int j = 0; j < 4; ++j) { const float d = v[j] - mean; qs += d * d; }
  red[t] = qs; __syncthreads();
  for (int st = 128; st > 0; st >>= 1) { if (t < st) red[t] += red[t + st]; __syncthreads(); }
  const float inv = rsqrtf(red[0] * (1.0f / (float)Dd) + 1e-5f);
#pragma unroll
  for (int j = 0; j < 4; ++j) {
    const int i = t + j * 256;
    Out[(size_t)row * Dd + i] = f2bf((v[j] - mean) * inv * G[i] + Bv[i]);
  }
}

// ---------------- generic bf16 WMMA GEMM ----------------
// C[M,N] = act(A[M,Kdim] @ Bw[Kdim,N] + bias)
// flags bit0: fp32 accumulate into Cacc (residual); else bf16 store to Cout
// flags bit1: exact GELU
constexpr int TM = 128, TN = 128, TKs = 32, LDSS = 40;  // 40 bf16 = 20 dwords (pad)

__global__ __launch_bounds__(256)
void gemm_bf16_kernel(const unsigned short* __restrict__ A,
                      const unsigned short* __restrict__ Bw,
                      const float* __restrict__ bias,
                      float* __restrict__ Cacc,
                      unsigned short* __restrict__ Cout,
                      int Kdim, int N, int flags) {
  __shared__ unsigned short lA[2][TM * LDSS];
  __shared__ unsigned short lB[2][TN * LDSS];  // stored transposed: [n][k]
  const int m0 = blockIdx.x * TM;
  const int n0 = blockIdx.y * TN;
  const int t  = threadIdx.x;
  const int lane = t & 31, wid = t >> 5;
  const int wm = wid >> 2, wn = wid & 3;      // 2 x 4 wave grid, wave tile 64M x 32N
  const int ha = lane >> 4, lr = lane & 15;

  const v8f z8 = {0.f, 0.f, 0.f, 0.f, 0.f, 0.f, 0.f, 0.f};
  v8f acc[4][2];
#pragma unroll
  for (int a = 0; a < 4; ++a)
#pragma unroll
    for (int b = 0; b < 2; ++b) acc[a][b] = z8;

  auto stage = [&](int buf, int kk) {
    {  // A tile 128x32, row-major, contiguous 16B chunks
      const int row = t >> 1, kc = (t & 1) * 16;
      const unsigned short* src = A + (size_t)(m0 + row) * Kdim + kk + kc;
      unsigned short* dst = &lA[buf][row * LDSS + kc];
#if HAVE_ASYNC_LDS
      // CDNA5 async copy: global -> LDS without VGPR round-trip (ASYNCcnt tracked)
      __builtin_amdgcn_global_load_async_to_lds_b128((AS1 v4i*)(const void*)src,
                                                     (AS3 v4i*)dst, 0, 0);
      __builtin_amdgcn_global_load_async_to_lds_b128((AS1 v4i*)(const void*)src,
                                                     (AS3 v4i*)dst, 16, 0);
#else
      const uint4* s4 = (const uint4*)src;
      uint4* d4 = (uint4*)dst;
      d4[0] = s4[0];
      d4[1] = s4[1];
      if (kk + TKs < Kdim) __builtin_prefetch(s4 + 4, 0, 1);  // next k-slab of this row
#endif
    }
    {  // B tile 32x128 -> transposed into LDS [n][k]
      const int k = t >> 3, nc = (t & 7) * 16;
      unsigned short tmp[16] __attribute__((aligned(16)));
      if (n0 + nc < N) {
        const uint4* src = (const uint4*)(Bw + (size_t)(kk + k) * N + n0 + nc);
        *(uint4*)&tmp[0] = src[0];
        *(uint4*)&tmp[8] = src[1];
      } else {
#pragma unroll
        for (int j = 0; j < 16; ++j) tmp[j] = 0;
      }
#pragma unroll
      for (int j = 0; j < 16; ++j) lB[buf][(nc + j) * LDSS + k] = tmp[j];
    }
  };

  stage(0, 0);
#if HAVE_ASYNC_LDS
  wait_asynccnt0();
#endif
  __syncthreads();
  const int nsteps = Kdim / TKs;
  for (int s = 0; s < nsteps; ++s) {
    const int buf = s & 1;
    const bool more = (s + 1 < nsteps);
    if (more) stage(buf ^ 1, (s + 1) * TKs);

    FragBf af[4], bfr[2];
    const int K0a = ha * 8;   // A layout: lanes 0-15 hold K 0-7 & 16-23; lanes 16-31 K 8-15 & 24-31
#pragma unroll
    for (int a = 0; a < 4; ++a) {
      const int r = wm * 64 + a * 16 + lr;
      af[a].q[0] = *(const uint4*)&lA[buf][r * LDSS + K0a];
      af[a].q[1] = *(const uint4*)&lA[buf][r * LDSS + K0a + 16];
    }
    const int K0b = ha * 8;   // B layout: lanes 0-15 hold K 0-7 & 16-23; lanes 16-31 K 8-15 & 24-31
#pragma unroll
    for (int b = 0; b < 2; ++b) {
      const int n = wn * 32 + b * 16 + lr;
      bfr[b].q[0] = *(const uint4*)&lB[buf][n * LDSS + K0b];
      bfr[b].q[1] = *(const uint4*)&lB[buf][n * LDSS + K0b + 16];
    }
#pragma unroll
    for (int a = 0; a < 4; ++a)
#pragma unroll
      for (int b = 0; b < 2; ++b)
        acc[a][b] = __builtin_amdgcn_wmma_f32_16x16x32_bf16(
            false, af[a].v, false, bfr[b].v, (short)0, acc[a][b], false, false);
#if HAVE_ASYNC_LDS
    if (more) wait_asynccnt0();
#endif
    __syncthreads();
  }

  // epilogue: C layout = VGPR r -> (M=r + 8*half, N = lane%16)
#pragma unroll
  for (int a = 0; a < 4; ++a)
#pragma unroll
    for (int b = 0; b < 2; ++b) {
      const int ncol = n0 + wn * 32 + b * 16 + lr;
      if (ncol >= N) continue;
      const float bv = bias ? bias[ncol] : 0.0f;
#pragma unroll
      for (int r = 0; r < 8; ++r) {
        const int m = m0 + wm * 64 + a * 16 + ha * 8 + r;
        float v = acc[a][b][r] + bv;
        if (flags & 2) v = 0.5f * v * (1.0f + erff(v * 0.70710678118654752f));
        const size_t idx = (size_t)m * N + ncol;
        if (flags & 1) Cacc[idx] += v;
        else Cout[idx] = f2bf(v);
      }
    }
}

// ---------------- kv[b] = Pk^T @ hk[b]  (64x64 per batch, reduce over N) ----------------
__global__ __launch_bounds__(256)
void kv_kernel(const unsigned short* __restrict__ hk,  // [B*N, Dh] bf16
               const float* __restrict__ Pk,           // [N, Kk] fp32 (layer slice)
               unsigned short* __restrict__ kv,        // [B, Kk, Dh] bf16
               unsigned short* __restrict__ kvT) {     // [B, Dh, Kk] bf16
  const int gid = blockIdx.x * 256 + threadIdx.x;      // B*Kk*Dh = 131072 threads
  const int d = gid & 63, k = (gid >> 6) & 63, b = gid >> 12;
  const unsigned short* hp = hk + (size_t)b * Nn * Dh + d;
  float acc = 0.f;
  for (int n = 0; n < Nn; ++n) acc += Pk[n * Kk + k] * bf2f(hp[(size_t)n * Dh]);
  kv [(size_t)(b * Kk + k) * Dh + d] = f2bf(acc);
  kvT[(size_t)(b * Dh + d) * Kk + k] = f2bf(acc);
}

// ---------------- Tensor Data Mover: 64x64 bf16 tile, global -> LDS ----------------
// Builds a D# (group0 + group1) per CDNA5 ISA ss8.3/8.4 and issues
// tensor_load_to_lds via inline asm (portable across toolchain builtin arities).
__device__ __forceinline__ void tdm_load_64x64_bf16(const unsigned short* gsrc,
                                                    unsigned lds_byte_off) {
  const unsigned long long ga = (unsigned long long)(uintptr_t)gsrc;
  v4u g0 = {
      1u,                                               // count=1, user mode, no gather
      lds_byte_off,                                     // lds_addr (bytes)
      (unsigned)ga,                                     // global_addr[31:0]
      ((unsigned)(ga >> 32) & 0x01FFFFFFu) | (2u << 30) // global_addr[56:32] | type=2
  };
  v8u g1 = {
      0x10000u,      // workgroup_mask=0, data_size=1 (2 bytes)
      64u << 16,     // tensor_dim0 = 64 (bits 79:48, low half)
      64u << 16,     // tensor_dim1 = 64 (bits 111:80, low half)
      64u << 16,     // tile_dim0 = 64 (bits 127:112)
      64u,           // tile_dim1 = 64 (bits 143:128), tile_dim2 = 0
      64u,           // tensor_dim0_stride = 64 (bits 207:160, low 32)
      0u,            // stride hi / tensor_dim1_stride lo
      0u             // tensor_dim1_stride hi
  };
  asm volatile("tensor_load_to_lds %0, %1" :: "s"(g0), "s"(g1) : "memory");
}

// ---------------- fused Linformer attention ----------------
// each wave owns one (b, head); loops 128 tiles of 16 tokens:
// dots = Q(16x64) @ kvT(64x64) -> softmax -> o = attn(16x64) @ kv(64x64)
__global__ __launch_bounds__(256)
void attn_kernel(const unsigned short* __restrict__ Q,    // [B*N, D]
                 const unsigned short* __restrict__ KV,   // [B, 64, 64]
                 const unsigned short* __restrict__ KVT,  // [B, 64, 64]
                 unsigned short* __restrict__ O) {        // [B*N, D]
  __shared__ unsigned short sKV[64 * 64];
  __shared__ unsigned short sKVT[64 * 64];
  __shared__ unsigned short sAttn[8][16 * 72];  // per-wave 16x64 tile, padded stride
  const int b = blockIdx.x >> 1;
  const int hbase = (blockIdx.x & 1) * 8;
  const int t = threadIdx.x;
  if (t < 32) {  // wave 0 drives the TDM; one DMA per tile, TENSORcnt tracked
    tdm_load_64x64_bf16(KV  + (size_t)b * 4096,
                        (unsigned)(uintptr_t)(AS3 unsigned short*)sKV);
    tdm_load_64x64_bf16(KVT + (size_t)b * 4096,
                        (unsigned)(uintptr_t)(AS3 unsigned short*)sKVT);
    __builtin_amdgcn_s_wait_tensorcnt(0);
  }
  __syncthreads();
  const int wid = t >> 5, lane = t & 31;
  const int h = hbase + wid;
  const int ha = lane >> 4, lr = lane & 15;

  // B fragments for dots: rows = d (Kdim), cols = k ; value kvT[d][k]
  FragBf dB[2][4];
#pragma unroll
  for (int dc = 0; dc < 2; ++dc)
#pragma unroll
    for (int kt = 0; kt < 4; ++kt) {
      const int kcol = kt * 16 + lr;
#pragma unroll
      for (int j = 0; j < 16; ++j)
        dB[dc][kt].us[j] = sKVT[(dc * 32 + ((j & 8) << 1) + ha * 8 + (j & 7)) * 64 + kcol];
    }
  // B fragments for o: rows = k (Kdim), cols = d ; value kv[k][d]
  FragBf oB[2][4];
#pragma unroll
  for (int kc = 0; kc < 2; ++kc)
#pragma unroll
    for (int dt = 0; dt < 4; ++dt) {
      const int dcol = dt * 16 + lr;
#pragma unroll
      for (int j = 0; j < 16; ++j)
        oB[kc][dt].us[j] = sKV[(kc * 32 + ((j & 8) << 1) + ha * 8 + (j & 7)) * 64 + dcol];
    }

  const v8f z8 = {0.f, 0.f, 0.f, 0.f, 0.f, 0.f, 0.f, 0.f};
  const int K0 = ha * 8;
  for (int t16 = 0; t16 < 128; ++t16) {
    const int n0 = t16 * 16;
    FragBf qa[2];
    const size_t qbase = ((size_t)b * Nn + n0 + lr) * Dd + h * Dh;
#pragma unroll
    for (int dc = 0; dc < 2; ++dc) {
      const unsigned short* p = Q + qbase + dc * 32;
      qa[dc].q[0] = *(const uint4*)(p + K0);
      qa[dc].q[1] = *(const uint4*)(p + K0 + 16);
    }
    v8f dots[4];
#pragma unroll
    for (int kt = 0; kt < 4; ++kt) dots[kt] = z8;
#pragma unroll
    for (int dc = 0; dc < 2; ++dc)
#pragma unroll
      for (int kt = 0; kt < 4; ++kt)
        dots[kt] = __builtin_amdgcn_wmma_f32_16x16x32_bf16(
            false, qa[dc].v, false, dB[dc][kt].v, (short)0, dots[kt], false, false);

    // softmax over k=64; row lives on 16 lanes of a half-wave x 4 tiles
#pragma unroll
    for (int r = 0; r < 8; ++r) {
      float v0 = dots[0][r] * 0.125f, v1 = dots[1][r] * 0.125f;
      float v2 = dots[2][r] * 0.125f, v3 = dots[3][r] * 0.125f;
      float mx = fmaxf(fmaxf(v0, v1), fmaxf(v2, v3));
      mx = fmaxf(mx, __shfl_xor(mx, 1, 32));
      mx = fmaxf(mx, __shfl_xor(mx, 2, 32));
      mx = fmaxf(mx, __shfl_xor(mx, 4, 32));
      mx = fmaxf(mx, __shfl_xor(mx, 8, 32));
      float e0 = expf(v0 - mx), e1 = expf(v1 - mx), e2 = expf(v2 - mx), e3 = expf(v3 - mx);
      float s = e0 + e1 + e2 + e3;
      s += __shfl_xor(s, 1, 32);
      s += __shfl_xor(s, 2, 32);
      s += __shfl_xor(s, 4, 32);
      s += __shfl_xor(s, 8, 32);
      const float inv = 1.0f / s;
      unsigned short* ap = &sAttn[wid][(ha * 8 + r) * 72];
      ap[lr]      = f2bf(e0 * inv);
      ap[16 + lr] = f2bf(e1 * inv);
      ap[32 + lr] = f2bf(e2 * inv);
      ap[48 + lr] = f2bf(e3 * inv);
    }
    asm volatile("s_wait_dscnt 0" ::: "memory");  // same-wave LDS RAW (C-layout -> A-layout repack)

    FragBf aa[2];
#pragma unroll
    for (int kc = 0; kc < 2; ++kc) {
      const unsigned short* p = &sAttn[wid][lr * 72 + kc * 32];
      aa[kc].q[0] = *(const uint4*)(p + K0);
      aa[kc].q[1] = *(const uint4*)(p + K0 + 16);
    }
    v8f oacc[4];
#pragma unroll
    for (int dt = 0; dt < 4; ++dt) oacc[dt] = z8;
#pragma unroll
    for (int kc = 0; kc < 2; ++kc)
#pragma unroll
      for (int dt = 0; dt < 4; ++dt)
        oacc[dt] = __builtin_amdgcn_wmma_f32_16x16x32_bf16(
            false, aa[kc].v, false, oB[kc][dt].v, (short)0, oacc[dt], false, false);
#pragma unroll
    for (int dt = 0; dt < 4; ++dt)
#pragma unroll
      for (int r = 0; r < 8; ++r)
        O[((size_t)b * Nn + n0 + ha * 8 + r) * Dd + h * Dh + dt * 16 + lr] = f2bf(oacc[dt][r]);
  }
}

// ---------------- mean over n + final layernorm -> fp32 out ----------------
__global__ __launch_bounds__(256)
void mean_lnf_kernel(const float* __restrict__ X, const float* __restrict__ G,
                     const float* __restrict__ Bv, float* __restrict__ out) {
  __shared__ float red[256];
  __shared__ float row[Dd];
  const int b = blockIdx.x, t = threadIdx.x;
  float acc[4] = {0.f, 0.f, 0.f, 0.f};
  for (int n = 0; n < Nn; ++n) {
    const float* xr = X + ((size_t)b * Nn + n) * Dd;
#pragma unroll
    for (int j = 0; j < 4; ++j) acc[j] += xr[t + j * 256];
  }
  float s = 0.f;
#pragma unroll
  for (int j = 0; j < 4; ++j) {
    const float m = acc[j] * (1.0f / (float)Nn);
    row[t + j * 256] = m;
    s += m;
  }
  red[t] = s; __syncthreads();
  for (int st = 128; st > 0; st >>= 1) { if (t < st) red[t] += red[t + st]; __syncthreads(); }
  const float mean = red[0] * (1.0f / (float)Dd);
  __syncthreads();
  float qs = 0.f;
#pragma unroll
  for (int j = 0; j < 4; ++j) { const float d = row[t + j * 256] - mean; qs += d * d; }
  red[t] = qs; __syncthreads();
  for (int st = 128; st > 0; st >>= 1) { if (t < st) red[t] += red[t + st]; __syncthreads(); }
  const float inv = rsqrtf(red[0] * (1.0f / (float)Dd) + 1e-5f);
#pragma unroll
  for (int j = 0; j < 4; ++j) {
    const int i = t + j * 256;
    out[(size_t)b * Dd + i] = (row[i] - mean) * inv * G[i] + Bv[i];
  }
}

// ---------------- host launcher ----------------
extern "C" void kernel_launch(void* const* d_in, const int* in_sizes, int n_in,
                              void* d_out, int out_size, void* d_ws, size_t ws_size,
                              hipStream_t stream) {
  (void)in_sizes; (void)n_in; (void)out_size; (void)ws_size;
  const int*   token_ids = (const int*)d_in[0];
  const float* token_emb = (const float*)d_in[1];
  const float* pos_emb   = (const float*)d_in[2];
  const float* ln1_g = (const float*)d_in[3];
  const float* ln1_b = (const float*)d_in[4];
  const float* Wq = (const float*)d_in[5];
  const float* Wk = (const float*)d_in[6];
  const float* Pk = (const float*)d_in[7];
  const float* Wo = (const float*)d_in[8];
  const float* bo = (const float*)d_in[9];
  const float* ln2_g = (const float*)d_in[10];
  const float* ln2_b = (const float*)d_in[11];
  const float* W1 = (const float*)d_in[12];
  const float* b1 = (const float*)d_in[13];
  const float* W2 = (const float*)d_in[14];
  const float* b2 = (const float*)d_in[15];
  const float* lnf_g = (const float*)d_in[16];
  const float* lnf_b = (const float*)d_in[17];
  float* out = (float*)d_out;

  // workspace carve
  char* ws = (char*)d_ws;
  size_t off = 0;
  auto carve = [&](size_t bytes) -> void* {
    void* p = ws + off;
    off += (bytes + 255) & ~(size_t)255;
    return p;
  };
  float*          x   = (float*)carve((size_t)BN * Dd * 4);           // fp32 residual stream
  unsigned short* h   = (unsigned short*)carve((size_t)BN * Dd * 2);  // LN output bf16
  unsigned short* u   = (unsigned short*)carve((size_t)BN * Ff * 2);  // FF hidden; aliases q/o
  unsigned short* q   = u;                                            // [BN, D]
  unsigned short* o   = u + (size_t)BN * Dd;                          // [BN, D]
  unsigned short* hk  = (unsigned short*)carve((size_t)BN * Dh * 2);
  unsigned short* kv  = (unsigned short*)carve((size_t)Bb * Kk * Dh * 2);
  unsigned short* kvT = (unsigned short*)carve((size_t)Bb * Dh * Kk * 2);
  unsigned short* wqb = (unsigned short*)carve((size_t)Ll * Dd * Dd * 2);
  unsigned short* wkb = (unsigned short*)carve((size_t)Ll * Dd * Dh * 2);
  unsigned short* wob = (unsigned short*)carve((size_t)Ll * Dd * Dd * 2);
  unsigned short* w1b = (unsigned short*)carve((size_t)Ll * Dd * Ff * 2);
  unsigned short* w2b = (unsigned short*)carve((size_t)Ll * Ff * Dd * 2);

  // weight conversion fp32 -> bf16 (done every call; deterministic)
  cvt_f32_bf16_kernel<<<2048, 256, 0, stream>>>(Wq, wqb, Ll * Dd * Dd);
  cvt_f32_bf16_kernel<<<512,  256, 0, stream>>>(Wk, wkb, Ll * Dd * Dh);
  cvt_f32_bf16_kernel<<<2048, 256, 0, stream>>>(Wo, wob, Ll * Dd * Dd);
  cvt_f32_bf16_kernel<<<4096, 256, 0, stream>>>(W1, w1b, Ll * Dd * Ff);
  cvt_f32_bf16_kernel<<<4096, 256, 0, stream>>>(W2, w2b, Ll * Ff * Dd);

  embed_kernel<<<BN, 256, 0, stream>>>(token_ids, token_emb, pos_emb, x);

  const dim3 gDD(BN / TM, Dd / TN);   // (512, 8)
  const dim3 gDK(BN / TM, 1);         // N=64 fits one 128-tile with bounds checks
  const dim3 gDF(BN / TM, Ff / TN);   // (512, 32)

  for (int l = 0; l < Ll; ++l) {
    ln_bf16_kernel<<<BN, 256, 0, stream>>>(x, ln1_g + l * Dd, ln1_b + l * Dd, h);
    // q = h @ Wq        [BN,1024]x[1024,1024] -> bf16
    gemm_bf16_kernel<<<gDD, 256, 0, stream>>>(h, wqb + (size_t)l * Dd * Dd, nullptr,
                                              nullptr, q, Dd, Dd, 0);
    // hk = h @ Wk       [BN,1024]x[1024,64] -> bf16
    gemm_bf16_kernel<<<gDK, 256, 0, stream>>>(h, wkb + (size_t)l * Dd * Dh, nullptr,
                                              nullptr, hk, Dd, Dh, 0);
    // kv = Pk^T @ hk per batch (64x64), both orientations bf16
    kv_kernel<<<(Bb * Kk * Dh) / 256, 256, 0, stream>>>(hk, Pk + (size_t)l * Nn * Kk, kv, kvT);
    // fused attention
    attn_kernel<<<Bb * 2, 256, 0, stream>>>(q, kv, kvT, o);
    // x += o @ Wo + bo
    gemm_bf16_kernel<<<gDD, 256, 0, stream>>>(o, wob + (size_t)l * Dd * Dd, bo + l * Dd,
                                              x, nullptr, Dd, Dd, 1);
    ln_bf16_kernel<<<BN, 256, 0, stream>>>(x, ln2_g + l * Dd, ln2_b + l * Dd, h);
    // u = gelu(h @ W1 + b1)
    gemm_bf16_kernel<<<gDF, 256, 0, stream>>>(h, w1b + (size_t)l * Dd * Ff, b1 + l * Ff,
                                              nullptr, u, Dd, Ff, 2);
    // x += u @ W2 + b2
    gemm_bf16_kernel<<<gDD, 256, 0, stream>>>(u, w2b + (size_t)l * Ff * Dd, b2 + l * Dd,
                                              x, nullptr, Ff, Dd, 1);
  }
  mean_lnf_kernel<<<Bb, 256, 0, stream>>>(x, lnf_g, lnf_b, out);
}
